// AudioLDM2_DDCM_2044404433534
// MI455X (gfx1250) — matrix operand order repaired and checked by reading, not compiled, not run on hardware
//
#include <hip/hip_runtime.h>
#include <hip/hip_bf16.h>
#include <math.h>

// ---------------------------------------------------------------------------
// Problem constants (match reference: B=32, K=1024, D=8*250*16=32000)
// ---------------------------------------------------------------------------
#define NB    32          // batch rows (M)
#define NK    1024        // codebook entries (N)
#define ND    32000       // flattened latent dim (reduction)
#define DC    640         // D-chunk per block (multiple of 64; 32000/640 = 50)
#define KSTEP 32          // WMMA bf16 K depth
#define WAVES 4           // waves per block
#define NT_BLOCK (WAVES * 16)  // 64 codes per block

typedef __attribute__((ext_vector_type(16))) __bf16 v16bf;
typedef __attribute__((ext_vector_type(8)))  float  v8f;

union Frag16 {            // 16 bf16 per lane = 8 VGPRs
    v16bf  v;
    uint4  q[2];
    __bf16 h[16];
};
union Acc8 {              // 8 f32 per lane = 8 VGPRs (16x16 f32 C/D)
    v8f   v;
    float f[8];
};

// ---------------------------------------------------------------------------
// Kernel 0: zero the partial-dot accumulator (32*1024 f32)
// ---------------------------------------------------------------------------
__global__ __launch_bounds__(256) void vq_zero(float* __restrict__ dot) {
    int i = blockIdx.x * 256 + threadIdx.x;
    if (i < NB * NK) dot[i] = 0.0f;
}

// ---------------------------------------------------------------------------
// Kernel 1: row norms. blocks 0..31 -> ||x_b||^2 ; blocks 32..1055 -> ||c_k||^2
// ---------------------------------------------------------------------------
__global__ __launch_bounds__(256) void vq_norms(const float* __restrict__ lat,
                                                const float* __restrict__ cb,
                                                float* __restrict__ norms) {
    __shared__ float red[256];
    int row = blockIdx.x;
    const float* p = (row < NB) ? (lat + (size_t)row * ND)
                                : (cb  + (size_t)(row - NB) * ND);
    float s = 0.0f;
    for (int i = threadIdx.x; i < ND; i += 256) {
        float v = p[i];
        s = fmaf(v, v, s);
    }
    red[threadIdx.x] = s;
    __syncthreads();
    for (int off = 128; off > 0; off >>= 1) {
        if (threadIdx.x < off) red[threadIdx.x] += red[threadIdx.x + off];
        __syncthreads();
    }
    if (threadIdx.x == 0) norms[row] = red[0];
}

// ---------------------------------------------------------------------------
// Kernel 2: WMMA GEMM partials.  dot[b,k] += sum_{d in chunk} x[b,d]*c[k,d]
// grid.x = NK/NT_BLOCK code tiles, grid.y = ND/DC reduction splits (16 x 50).
// ---------------------------------------------------------------------------
__global__ __launch_bounds__(WAVES * 32) void vq_gemm(const float* __restrict__ lat,
                                                      const float* __restrict__ cb,
                                                      float* __restrict__ dot) {
    __shared__ __bf16 sA[NB * DC];   // latents chunk as bf16 (40 KB of 320 KB WGP LDS)

    const int nBase = blockIdx.x * NT_BLOCK;
    const int d0    = blockIdx.y * DC;
    const int tid   = threadIdx.x;

    // ---- stage latents [32 x DC] slice into LDS as bf16 (row-major) ----
    // native RNE cvt: lowers to v_cvt_pk_bf16_f32
    for (int i = tid; i < NB * DC; i += WAVES * 32) {
        int r = i / DC, c = i - r * DC;
        sA[i] = (__bf16)lat[(size_t)r * ND + d0 + c];
    }
    __syncthreads();

    const int wave   = tid >> 5;
    const int lane   = tid & 31;
    const int lane16 = lane & 15;
    const int hi     = lane >> 4;          // 0 = lanes 0-15, 1 = lanes 16-31

    // B layout (32x16 bf16): lane n holds column N=n, K=0..15 (lo) / 16..31 (hi)
    const int code = nBase + wave * 16 + lane16;
    const float* __restrict__ brow = cb + (size_t)code * ND + d0 + (hi << 4);

    // A layout (16x32 bf16): lane<16 -> K 0..15; lane>=16 -> K 16..31
    const int kOffLo = hi ? 16 : 0;

    Acc8 acc0; Acc8 acc1;
    acc0.v = (v8f){0.f,0.f,0.f,0.f,0.f,0.f,0.f,0.f};
    acc1.v = (v8f){0.f,0.f,0.f,0.f,0.f,0.f,0.f,0.f};

    #pragma unroll 2
    for (int kk = 0; kk < DC; kk += KSTEP) {
        // --- A fragments from LDS (two aligned 16B reads per M-tile) ---
        Frag16 a0, a1;
        const __bf16* r0 = &sA[(size_t)lane16 * DC + kk];
        const __bf16* r1 = &sA[(size_t)(lane16 + 16) * DC + kk];
        a0.q[0] = *(const uint4*)(r0 + kOffLo);
        a0.q[1] = *(const uint4*)(r0 + 8 + kOffLo);
        a1.q[0] = *(const uint4*)(r1 + kOffLo);
        a1.q[1] = *(const uint4*)(r1 + 8 + kOffLo);

        // --- B fragment: 16 consecutive f32 from codebook row, native cvt ---
        const float* bp = brow + kk;
        __builtin_prefetch(bp + KSTEP, 0, 0);   // global_prefetch next K-step
        Frag16 b;
        #pragma unroll
        for (int q = 0; q < 4; ++q) {
            float4 f = ((const float4*)bp)[q];
            b.h[q * 4 + 0] = (__bf16)f.x;
            b.h[q * 4 + 1] = (__bf16)f.y;
            b.h[q * 4 + 2] = (__bf16)f.z;
            b.h[q * 4 + 3] = (__bf16)f.w;
        }

        // --- two 16x16x32 bf16 WMMAs (M-tile 0-15 and 16-31) ---
        acc0.v = __builtin_amdgcn_wmma_f32_16x16x32_bf16(
            false, a0.v, false, b.v, (short)0, acc0.v, false, false);
        acc1.v = __builtin_amdgcn_wmma_f32_16x16x32_bf16(
            false, a1.v, false, b.v, (short)0, acc1.v, false, false);
    }

    // ---- C/D layout: VGPR r -> M = r (lanes 0-15) / M = 8+r (lanes 16-31) ----
    const int rbase = hi * 8;
    #pragma unroll
    for (int r = 0; r < 8; ++r) {
        atomicAdd(&dot[(rbase + r)      * NK + code], acc0.f[r]);
        atomicAdd(&dot[(16 + rbase + r) * NK + code], acc1.f[r]);
    }
}

// ---------------------------------------------------------------------------
// Kernel 3: finalize — argmin per row, indices / min-dist / usage outputs
// out layout (floats): idx[32] | quant[32*32000] | mind[32] | usage[1024]
// ---------------------------------------------------------------------------
__global__ __launch_bounds__(256) void vq_finalize(const float* __restrict__ norms,
                                                   const float* __restrict__ dot,
                                                   const float* __restrict__ usage_in,
                                                   float* __restrict__ out,
                                                   int* __restrict__ ws_idx) {
    const float* x2 = norms;
    const float* c2 = norms + NB;
    float* out_idx   = out;
    float* out_mind  = out + NB + (size_t)NB * ND;
    float* out_usage = out_mind + NB;

    __shared__ float sval[256];
    __shared__ int   sidx[256];
    __shared__ int   chosen[NB];

    for (int k = threadIdx.x; k < NK; k += 256) out_usage[k] = usage_in[k];

    for (int b = 0; b < NB; ++b) {
        float best = 3.4e38f;
        int   bi   = 0;
        for (int k = threadIdx.x; k < NK; k += 256) {
            float d2 = x2[b] + c2[k] - 2.0f * dot[b * NK + k];
            if (d2 < best) { best = d2; bi = k; }
        }
        sval[threadIdx.x] = best;
        sidx[threadIdx.x] = bi;
        __syncthreads();
        for (int off = 128; off > 0; off >>= 1) {
            if (threadIdx.x < off) {
                float v = sval[threadIdx.x + off];
                int   i = sidx[threadIdx.x + off];
                if (v < sval[threadIdx.x] ||
                    (v == sval[threadIdx.x] && i < sidx[threadIdx.x])) {
                    sval[threadIdx.x] = v;
                    sidx[threadIdx.x] = i;
                }
            }
            __syncthreads();
        }
        if (threadIdx.x == 0) {
            int idx = sidx[0];
            chosen[b] = idx;
            ws_idx[b] = idx;
            out_idx[b]  = (float)idx;
            out_mind[b] = sqrtf(fmaxf(sval[0], 0.0f));
        }
        __syncthreads();
    }

    if (threadIdx.x == 0) {
        for (int b = 0; b < NB; ++b) out_usage[chosen[b]] += 1.0f;
    }
}

// ---------------------------------------------------------------------------
// Kernel 4: gather quantized = codebook[idx[b]]  (L2-resident rows)
// ---------------------------------------------------------------------------
__global__ __launch_bounds__(256) void vq_gather(const float* __restrict__ cb,
                                                 const int* __restrict__ ws_idx,
                                                 float* __restrict__ out_q) {
    int i = blockIdx.x * 256 + threadIdx.x;
    if (i < NB * ND) {
        int b = i / ND;
        int j = i - b * ND;
        out_q[i] = cb[(size_t)ws_idx[b] * ND + j];
    }
}

// ---------------------------------------------------------------------------
// Host launcher
// ---------------------------------------------------------------------------
extern "C" void kernel_launch(void* const* d_in, const int* in_sizes, int n_in,
                              void* d_out, int out_size, void* d_ws, size_t ws_size,
                              hipStream_t stream) {
    (void)in_sizes; (void)n_in; (void)out_size; (void)ws_size;

    const float* lat   = (const float*)d_in[0];   // [32, 8,250,16]
    const float* cb    = (const float*)d_in[1];   // [1024, 8,250,16]
    const float* usage = (const float*)d_in[2];   // [1024]

    float* ws    = (float*)d_ws;
    float* norms = ws;                        // 1056 floats
    float* dot   = ws + (NB + NK);            // 32*1024 floats
    int*   idxws = (int*)(ws + (NB + NK) + NB * NK);  // 32 ints

    float* out = (float*)d_out;

    vq_zero<<<(NB * NK + 255) / 256, 256, 0, stream>>>(dot);
    vq_norms<<<NB + NK, 256, 0, stream>>>(lat, cb, norms);
    vq_gemm<<<dim3(NK / NT_BLOCK, ND / DC), WAVES * 32, 0, stream>>>(lat, cb, dot);
    vq_finalize<<<1, 256, 0, stream>>>(norms, dot, usage, out, idxws);
    vq_gather<<<(NB * ND + 255) / 256, 256, 0, stream>>>(cb, idxws, out + NB);
}
